// MultiHeadAttention_ch_20624432955932
// MI455X (gfx1250) — matrix operand order repaired and checked by reading, not compiled, not run on hardware
//
#include <hip/hip_runtime.h>
#include <hip/hip_bf16.h>

// ---------------------------------------------------------------------------
// Types for CDNA5 WMMA (wave32): v_wmma_f32_16x16x32_bf16
// ---------------------------------------------------------------------------
typedef __attribute__((ext_vector_type(16))) __bf16 v16bf;
typedef __attribute__((ext_vector_type(8)))  __bf16 bf16x8;
typedef __attribute__((ext_vector_type(4)))  __bf16 bf16x4;
typedef __attribute__((ext_vector_type(8)))  float  v8f;
typedef __attribute__((ext_vector_type(4)))  float  v4f;

union FragAB { v16bf v; bf16x8 h[2]; };

// Problem constants
#define BB 8
#define LL 1024
#define DD 512
#define HH 8
#define QD 64
#define MM (BB * LL)          // 8192
#define KK 512

// ---------------------------------------------------------------------------
// fp32 -> bf16 conversion (vectorized: 16B in / 8B out per lane)
// ---------------------------------------------------------------------------
__global__ void f32_to_bf16_kernel(const v4f* __restrict__ in,
                                   bf16x4* __restrict__ out, int n4) {
    int i = blockIdx.x * blockDim.x + threadIdx.x;
    if (i < n4) {
        v4f x = in[i];
        bf16x4 o;
#pragma unroll
        for (int e = 0; e < 4; ++e) o[e] = (__bf16)x[e];
        out[i] = o;
    }
}

// ---------------------------------------------------------------------------
// Generic WMMA GEMM:  C[M,N] = A[M,K](bf16,row-major) * W[N,K](bf16)^T + bias
// Block: 256 threads = 8 waves; block tile 128(M) x 64(N); K = 512.
// The 64x512 bf16 weight panel is staged into LDS once per block with
// GLOBAL_LOAD_ASYNC_TO_LDS_B128 (ASYNCcnt) so all 8 waves read it from LDS.
// MODE 0: bf16 row-major [M,N]
// MODE 1: bf16 head-major  out[((b*H+h)*L+pos)*64 + d]   (Q / K)
// MODE 2: bf16 d-major     out[((b*H+h)*64+d)*L + pos]   (V transposed)
// MODE 3: fp32 row-major [M,N]                            (final out)
// ---------------------------------------------------------------------------
#define WKP 520   // padded LDS row stride (bf16 elems) for the weight panel

template <int MODE>
__global__ void gemm_bf16_wmma(const __bf16* __restrict__ A,
                               const __bf16* __restrict__ W,
                               const float*  __restrict__ bias,
                               void* __restrict__ out,
                               int M, int N) {
    extern __shared__ char smem[];
    __bf16* Wl = (__bf16*)smem;                     // [64][WKP] bf16 panel

    const int tid  = threadIdx.x;
    const int lane = tid & 31;
    const int wave = tid >> 5;
    const int half = lane >> 4;     // 0/1
    const int l16  = lane & 15;

    const int mtile = blockIdx.x * 128 + wave * 16;
    const int ntile = blockIdx.y * 64;

    // ---- async stage of the W panel into LDS (64 rows x 512 bf16) ----
    {
        const unsigned ldsbase = __builtin_amdgcn_groupstaticsize(); // dyn-LDS base (0)
        const __bf16* wpan = W + (size_t)ntile * KK;
        // 64 rows * 64 chunks(16B) = 4096 chunks; 16 per thread
#pragma unroll
        for (int i = 0; i < 16; ++i) {
            const int idx = i * 256 + tid;
            const int row = idx >> 6;
            const int ch  = idx & 63;
            unsigned long long gaddr =
                (unsigned long long)(wpan + (size_t)row * KK + ch * 8);
            unsigned ldsoff = ldsbase + (unsigned)((row * WKP + ch * 8) * 2);
            asm volatile("global_load_async_to_lds_b128 %0, %1, off"
                         :: "v"(ldsoff), "v"(gaddr) : "memory");
        }
        asm volatile("s_wait_asynccnt 0x0" ::: "memory");
    }
    __syncthreads();

    const __bf16* arow = A + (size_t)(mtile + l16) * KK;
    const int akoff = half * 8;     // A-fragment K pattern
    const int bkoff = half * 8;     // B-fragment K pattern

    v8f acc[4] = {};

    for (int k0 = 0; k0 < KK; k0 += 32) {
        FragAB a;
        a.h[0] = *(const bf16x8*)(arow + k0 + akoff);
        a.h[1] = *(const bf16x8*)(arow + k0 + akoff + 16);
#pragma unroll
        for (int nn = 0; nn < 4; ++nn) {
            const __bf16* wr = Wl + (nn * 16 + l16) * WKP + k0 + bkoff;
            FragAB b;
            b.h[0] = *(const bf16x8*)(wr);
            b.h[1] = *(const bf16x8*)(wr + 16);
            acc[nn] = __builtin_amdgcn_wmma_f32_16x16x32_bf16(
                false, a.v, false, b.v, (short)0, acc[nn], false, false);
        }
    }

    const int mrow0 = mtile + half * 8;
#pragma unroll
    for (int nn = 0; nn < 4; ++nn) {
        const int n  = ntile + nn * 16 + l16;
        const float bv = bias[n];
#pragma unroll
        for (int r = 0; r < 8; ++r) {
            const int m = mrow0 + r;
            const float val = acc[nn][r] + bv;
            if (MODE == 3) {
                ((float*)out)[(size_t)m * N + n] = val;
            } else {
                const __bf16 o = (__bf16)val;
                if (MODE == 0) {
                    ((__bf16*)out)[(size_t)m * N + n] = o;
                } else {
                    const int b = m >> 10, pos = m & (LL - 1);
                    const int h = n >> 6,  d   = n & (QD - 1);
                    if (MODE == 1)
                        ((__bf16*)out)[(((size_t)(b * HH + h) * LL + pos) << 6) + d] = o;
                    else // MODE 2
                        ((__bf16*)out)[(((size_t)(b * HH + h) * QD + d) << 10) + pos] = o;
                }
            }
        }
    }
}

// ---------------------------------------------------------------------------
// Fused attention:  S = Q*K^T/8, softmax rows in LDS, stream fp32 scores to
// HBM with non-temporal b128 stores, write normalized probs as bf16 into an
// LDS P-buffer, then P*V with WMMA into merged[b,l,d*H+h] (bf16).
// One block = 128 threads (4 waves) per (b,h,16-query) row block.
// ---------------------------------------------------------------------------
#define SSTRIDE 1028            // fp32 score row stride (bank-conflict pad)
#define PSTRIDE 1032            // bf16 prob row stride  (bank-conflict pad)
#define S_BYTES (16 * SSTRIDE * 4)
#define P_BYTES (16 * PSTRIDE * 2)

__global__ void attention_fused_kernel(const __bf16* __restrict__ Qh,   // [BH,L,64]
                                       const __bf16* __restrict__ Kh,   // [BH,L,64]
                                       const __bf16* __restrict__ Vt,   // [BH,64,L]
                                       float* __restrict__ scores,      // [BH,L,L]
                                       __bf16* __restrict__ merged) {   // [B,L,512]
    extern __shared__ char smem[];
    float*  S      = (float*)smem;                          // 16 x SSTRIDE f32
    __bf16* P      = (__bf16*)(smem + S_BYTES);             // 16 x PSTRIDE bf16
    float*  part   = (float*)(smem + S_BYTES + P_BYTES);    // 8 x 16
    float*  rowmax = part + 128;                            // 16
    float*  rowsum = rowmax + 16;                           // 16

    const int tid  = threadIdx.x;       // 0..127
    const int lane = tid & 31;
    const int wave = tid >> 5;          // 0..3
    const int half = lane >> 4;
    const int l16  = lane & 15;

    const int bh    = blockIdx.x >> 6;  // 64 q-blocks per (b,h)
    const int qblk  = blockIdx.x & 63;
    const int qbase = qblk * 16;

    const __bf16* Qb = Qh + (size_t)bh * LL * QD;
    const __bf16* Kb = Kh + (size_t)bh * LL * QD;
    const __bf16* Vb = Vt + (size_t)bh * QD * LL;

    // Q fragments (16 queries x 64 dims = two K-steps of 32), per wave
    FragAB qa[2];
    {
        const __bf16* qrow = Qb + (size_t)(qbase + l16) * QD;
#pragma unroll
        for (int s = 0; s < 2; ++s) {
            const int kb = s * 32 + half * 8;
            qa[s].h[0] = *(const bf16x8*)(qrow + kb);
            qa[s].h[1] = *(const bf16x8*)(qrow + kb + 16);
        }
    }

    // ---- phase 1: S tiles (each wave does 16 of the 64 j-tiles) ----
    for (int jt = wave; jt < 64; jt += 4) {
        const __bf16* krow = Kb + (size_t)(jt * 16 + l16) * QD;
        v8f c = {};
#pragma unroll
        for (int s = 0; s < 2; ++s) {
            FragAB b;  // B = K^T: column n = K row (jt*16+n), K-dim contiguous
            b.h[0] = *(const bf16x8*)(krow + s * 32 + half * 8);
            b.h[1] = *(const bf16x8*)(krow + s * 32 + half * 8 + 16);
            c = __builtin_amdgcn_wmma_f32_16x16x32_bf16(
                false, qa[s].v, false, b.v, (short)0, c, false, false);
        }
#pragma unroll
        for (int r = 0; r < 8; ++r) {
            const int m = r + half * 8;
            S[m * SSTRIDE + jt * 16 + l16] = c[r] * 0.125f; // 1/sqrt(64)
        }
    }
    __syncthreads();

    // ---- softmax statistics: row = tid&15, chunk of 128 cols = tid>>4 ----
    const int srow  = tid & 15;
    const int chunk = tid >> 4;
    {
        float mx = -3.4e38f;
        const v4f* sr = (const v4f*)(S + srow * SSTRIDE + chunk * 128);
        for (int i = 0; i < 32; ++i) {
            v4f v = sr[i];
            mx = fmaxf(mx, fmaxf(fmaxf(v[0], v[1]), fmaxf(v[2], v[3])));
        }
        part[chunk * 16 + srow] = mx;
    }
    __syncthreads();
    if (tid < 16) {
        float mx = part[tid];
        for (int c = 1; c < 8; ++c) mx = fmaxf(mx, part[c * 16 + tid]);
        rowmax[tid] = mx;
    }
    __syncthreads();
    {
        const float mx = rowmax[srow];
        float sm = 0.f;
        const v4f* sr = (const v4f*)(S + srow * SSTRIDE + chunk * 128);
        for (int i = 0; i < 32; ++i) {
            v4f v = sr[i];
            sm += __expf(v[0] - mx) + __expf(v[1] - mx) +
                  __expf(v[2] - mx) + __expf(v[3] - mx);
        }
        part[chunk * 16 + srow] = sm;
    }
    __syncthreads();
    if (tid < 16) {
        float sm = 0.f;
        for (int c = 0; c < 8; ++c) sm += part[c * 16 + tid];
        rowsum[tid] = 1.0f / sm;
    }
    __syncthreads();

    // ---- normalize: NT-stream fp32 scores to HBM (b128) + bf16 -> LDS P ----
    {
        const int cb = tid * 8;     // 8 contiguous cols per thread
        for (int r = 0; r < 16; ++r) {
            const float mx = rowmax[r], inv = rowsum[r];
            const v4f* sr = (const v4f*)(S + r * SSTRIDE + cb);
            const v4f s0 = sr[0], s1 = sr[1];
            v4f e0, e1;
#pragma unroll
            for (int e = 0; e < 4; ++e) e0[e] = __expf(s0[e] - mx) * inv;
#pragma unroll
            for (int e = 0; e < 4; ++e) e1[e] = __expf(s1[e] - mx) * inv;
            float* gout = scores + ((size_t)bh * LL + qbase + r) * LL + cb;
            __builtin_nontemporal_store(e0, (v4f*)gout);
            __builtin_nontemporal_store(e1, (v4f*)(gout + 4));
            bf16x8 pb;
#pragma unroll
            for (int e = 0; e < 4; ++e) pb[e] = (__bf16)e0[e];
#pragma unroll
            for (int e = 0; e < 4; ++e) pb[e + 4] = (__bf16)e1[e];
            *(bf16x8*)(P + r * PSTRIDE + cb) = pb;
        }
    }
    __syncthreads();

    // ---- phase 2: atten = P * V ; each wave owns one 16-wide d tile ----
    {
        const int dblk = wave * 16;
        const __bf16* vrow = Vb + (size_t)(dblk + l16) * LL;  // Vt row = dim d
        const __bf16* prow = P + l16 * PSTRIDE + half * 8;    // A-pattern base
        v8f c0 = {}, c1 = {};
        for (int jb = 0; jb < LL; jb += 64) {
            FragAB a0, a1, b0, b1;
            a0.h[0] = *(const bf16x8*)(prow + jb);
            a0.h[1] = *(const bf16x8*)(prow + jb + 16);
            b0.h[0] = *(const bf16x8*)(vrow + jb + half * 8);
            b0.h[1] = *(const bf16x8*)(vrow + jb + half * 8 + 16);
            a1.h[0] = *(const bf16x8*)(prow + jb + 32);
            a1.h[1] = *(const bf16x8*)(prow + jb + 48);
            b1.h[0] = *(const bf16x8*)(vrow + jb + 32 + half * 8);
            b1.h[1] = *(const bf16x8*)(vrow + jb + 32 + half * 8 + 16);
            c0 = __builtin_amdgcn_wmma_f32_16x16x32_bf16(
                false, a0.v, false, b0.v, (short)0, c0, false, false);
            c1 = __builtin_amdgcn_wmma_f32_16x16x32_bf16(
                false, a1.v, false, b1.v, (short)0, c1, false, false);
        }
        // merged[b, q, d*H + h]  (the reference's (d,h)-major flatten)
        const int b = bh >> 3, h = bh & 7;
#pragma unroll
        for (int r = 0; r < 8; ++r) {
            const int q = qbase + r + half * 8;
            const int d = dblk + l16;
            merged[((size_t)(b * LL + q)) * (HH * QD) + d * HH + h] =
                (__bf16)(c0[r] + c1[r]);
        }
    }
}

// ---------------------------------------------------------------------------
// Host-side orchestration
// ---------------------------------------------------------------------------
extern "C" void kernel_launch(void* const* d_in, const int* in_sizes, int n_in,
                              void* d_out, int out_size, void* d_ws, size_t ws_size,
                              hipStream_t stream) {
    const float* x  = (const float*)d_in[0];
    const float* Wc = (const float*)d_in[1];
    const float* bc = (const float*)d_in[2];
    const float* Wq = (const float*)d_in[3];
    const float* bq = (const float*)d_in[4];
    const float* Wk = (const float*)d_in[5];
    const float* bk = (const float*)d_in[6];
    const float* Wv = (const float*)d_in[7];
    const float* bv = (const float*)d_in[8];
    const float* Wo = (const float*)d_in[9];
    const float* bo = (const float*)d_in[10];

    float* out    = (float*)d_out;                       // [8192, 512]
    float* scores = out + (size_t)MM * DD;               // [64, 1024, 1024]

    // Workspace carve-up (bf16 tensors)
    char* ws = (char*)d_ws;
    const size_t SZ_X  = (size_t)MM * DD * sizeof(__bf16);   // 8 MiB each
    const size_t SZ_W  = (size_t)DD * DD * sizeof(__bf16);   // 512 KiB each
    __bf16* xb     = (__bf16*)(ws);
    __bf16* xrb    = (__bf16*)(ws + SZ_X);
    __bf16* Qh     = (__bf16*)(ws + 2 * SZ_X);
    __bf16* Kh     = (__bf16*)(ws + 3 * SZ_X);
    __bf16* Vt     = (__bf16*)(ws + 4 * SZ_X);
    __bf16* merged = (__bf16*)(ws + 5 * SZ_X);
    __bf16* Wcb    = (__bf16*)(ws + 6 * SZ_X);
    __bf16* Wqb    = (__bf16*)(ws + 6 * SZ_X + SZ_W);
    __bf16* Wkb    = (__bf16*)(ws + 6 * SZ_X + 2 * SZ_W);
    __bf16* Wvb    = (__bf16*)(ws + 6 * SZ_X + 3 * SZ_W);
    __bf16* Wob    = (__bf16*)(ws + 6 * SZ_X + 4 * SZ_W);

    const int NX4 = MM * DD / 4;     // 1,048,576 float4 groups
    const int NW4 = DD * DD / 4;     // 65,536

    f32_to_bf16_kernel<<<(NX4 + 255) / 256, 256, 0, stream>>>((const v4f*)x,  (bf16x4*)xb,  NX4);
    f32_to_bf16_kernel<<<(NW4 + 255) / 256, 256, 0, stream>>>((const v4f*)Wc, (bf16x4*)Wcb, NW4);
    f32_to_bf16_kernel<<<(NW4 + 255) / 256, 256, 0, stream>>>((const v4f*)Wq, (bf16x4*)Wqb, NW4);
    f32_to_bf16_kernel<<<(NW4 + 255) / 256, 256, 0, stream>>>((const v4f*)Wk, (bf16x4*)Wkb, NW4);
    f32_to_bf16_kernel<<<(NW4 + 255) / 256, 256, 0, stream>>>((const v4f*)Wv, (bf16x4*)Wvb, NW4);
    f32_to_bf16_kernel<<<(NW4 + 255) / 256, 256, 0, stream>>>((const v4f*)Wo, (bf16x4*)Wob, NW4);

    dim3 ggrid(MM / 128, DD / 64);   // (64, 8)
    const size_t gemm_lds = (size_t)64 * WKP * sizeof(__bf16);   // 66,560 B
    // x_r = x*Wc^T + bc    (row-major bf16)
    gemm_bf16_wmma<0><<<ggrid, 256, gemm_lds, stream>>>(xb,  Wcb, bc, (void*)xrb, MM, DD);
    // Q = x*Wq^T + bq      (head-major)
    gemm_bf16_wmma<1><<<ggrid, 256, gemm_lds, stream>>>(xb,  Wqb, bq, (void*)Qh,  MM, DD);
    // K = x_r*Wk^T + bk    (head-major)
    gemm_bf16_wmma<1><<<ggrid, 256, gemm_lds, stream>>>(xrb, Wkb, bk, (void*)Kh,  MM, DD);
    // V = x_r*Wv^T + bv    (d-major transposed for B-fragment loads)
    gemm_bf16_wmma<2><<<ggrid, 256, gemm_lds, stream>>>(xrb, Wvb, bv, (void*)Vt,  MM, DD);

    // fused scores + softmax + P*V : one block per (b,h,16-query block)
    const int att_blocks = BB * HH * (LL / 16);          // 4096
    const size_t att_lds = (size_t)S_BYTES + P_BYTES + (128 + 16 + 16) * sizeof(float);
    attention_fused_kernel<<<att_blocks, 128, att_lds, stream>>>(Qh, Kh, Vt, scores, merged);

    // out = merged*Wo^T + bo  (fp32)
    gemm_bf16_wmma<3><<<ggrid, 256, gemm_lds, stream>>>(merged, Wob, bo, (void*)out, MM, DD);
}
